// LatentDynamicsNDE_18923625906471
// MI455X (gfx1250) — compile-verified, matrix operand order checked by reading, not run on hardware
//
#include <hip/hip_runtime.h>
#include <hip/hip_bf16.h>

typedef __attribute__((ext_vector_type(16))) __bf16 bf16x16;
typedef __attribute__((ext_vector_type(8)))  __bf16 bf16x8;
typedef __attribute__((ext_vector_type(8)))  float  f32x8;

#define BATCH    16384
#define ROW_F32  (28 * 256)   // trajectory row stride (floats)
#define LATENT   128
#define HID      256
#define NSTEP    54
#define DT       0.5f

#define FRAG_ELEMS 512        // bf16 elements per 16x16x32 B-fragment (32 lanes * 16)

// d_ws layout (bf16 element offsets)
#define OFF_W1   0            // 16*4  = 64  frags
#define OFF_W2   32768        // 16*8  = 128 frags
#define OFF_W3   98304        // 8*8   = 64  frags
#define OFF_WE1  131072       // 16*24 = 384 frags
#define OFF_WE2  327680       // 8*8   = 64  frags
#define OFF_WC1  360448       // 16*4  = 64  frags  (end = 393216 elems = 768 KB)

// LDS layout
#define LDS_W_BYTES   262144          // W1+W2+W3 fragments
#define ACT_STRIDE_Z  136             // K=128 + pad 8  (bank-conflict-free)
#define ACT_STRIDE_H  264             // K=256 + pad 8
#define ACT_ELEMS     (16 * ACT_STRIDE_H)   // 4224 bf16 = 8448 B per wave
#define WAVES_PER_BLK 4
#define ROWS_PER_BLK  (WAVES_PER_BLK * 16)
#define LDS_TOTAL     (LDS_W_BYTES + WAVES_PER_BLK * ACT_ELEMS * 2)  // 295936 B

__device__ __forceinline__ f32x8 wmma_bf16(bf16x16 a, bf16x16 b, f32x8 c) {
  return __builtin_amdgcn_wmma_f32_16x16x32_bf16(false, a, false, b, (short)0, c,
                                                 false, false);
}

// B-fragment: chunk-major layout. chunk0 (VGPR0-3 ks) at +0, chunk1 at +512B.
__device__ __forceinline__ bf16x16 load_bfrag(const __bf16* frag, int lane) {
  bf16x8 lo = *(const bf16x8*)(frag + lane * 8);
  bf16x8 hi = *(const bf16x8*)(frag + 256 + lane * 8);
  bf16x16 r;
#pragma unroll
  for (int i = 0; i < 8; ++i) { r[i] = lo[i]; r[i + 8] = hi[i]; }
  return r;
}

// A-fragment from row-major bf16 activation tile (16 rows x stride)
__device__ __forceinline__ bf16x16 load_afrag(const __bf16* act, int stride,
                                              int kt, int lane) {
  int m  = lane & 15;
  int kb = kt * 32 + ((lane >> 4) << 3);
  const __bf16* p = act + m * stride + kb;
  bf16x8 lo = *(const bf16x8*)(p);
  bf16x8 hi = *(const bf16x8*)(p + 16);
  bf16x16 r;
#pragma unroll
  for (int i = 0; i < 8; ++i) { r[i] = lo[i]; r[i + 8] = hi[i]; }
  return r;
}

// A-fragment straight from global f32 rows (encoder input)
__device__ __forceinline__ bf16x16 load_afrag_gf32(const float* base, int rowbase,
                                                   int stride, int kt, int lane) {
  int m  = lane & 15;
  int kb = kt * 32 + ((lane >> 4) << 3);
  const float* p = base + (size_t)(rowbase + m) * stride + kb;
  bf16x16 r;
#pragma unroll
  for (int i = 0; i < 8; ++i) { r[i] = (__bf16)p[i]; r[i + 8] = (__bf16)p[16 + i]; }
  return r;
}

// C-layout f32 tile -> row-major bf16 in LDS
__device__ __forceinline__ void store_tile_bf16(__bf16* act, int stride, int j,
                                                int lane, f32x8 v) {
  int n  = j * 16 + (lane & 15);
  int mb = (lane >> 4) << 3;
#pragma unroll
  for (int r = 0; r < 8; ++r) act[(mb + r) * stride + n] = (__bf16)v[r];
}

// Native v_tanh_f32 on CDNA5 when available (single TRANS op, co-executes
// with the WMMA/XDL pipe); guarded fallback via exp otherwise.
__device__ __forceinline__ float tanh_f(float x) {
#if __has_builtin(__builtin_amdgcn_tanhf)
  return __builtin_amdgcn_tanhf(x);
#elif __has_builtin(__builtin_amdgcn_tanh_f32)
  return __builtin_amdgcn_tanh_f32(x);
#else
  float xc = fminf(fmaxf(x, -10.f), 10.f);
  float e  = __expf(2.f * xc);
  return (e - 1.f) * __builtin_amdgcn_rcpf(e + 1.f);
#endif
}

struct WaveCtx {
  const __bf16* w1;  // LDS fragment bases
  const __bf16* w2;
  const __bf16* w3;
  __bf16* act;       // per-wave LDS scratch
  int lane;
  float b1v[16], b2v[16];
  float b3v[8], wtev[8], btev[8];
};

// out = f(t, zin): 3-layer tanh MLP, all WMMA
__device__ __forceinline__ void ode_f(const WaveCtx& W, float t,
                                      const f32x8 zin[8], f32x8 out[8]) {
  const int lane = W.lane;
  // stage zt = z + t*wte + bte as bf16 (stride 136)
#pragma unroll
  for (int j = 0; j < 8; ++j) {
    float tb = t * W.wtev[j] + W.btev[j];
    f32x8 zt;
#pragma unroll
    for (int r = 0; r < 8; ++r) zt[r] = zin[j][r] + tb;
    store_tile_bf16(W.act, ACT_STRIDE_Z, j, lane, zt);
  }
  // layer 1: K=128, N=256
  bf16x16 a4[4];
#pragma unroll
  for (int kt = 0; kt < 4; ++kt) a4[kt] = load_afrag(W.act, ACT_STRIDE_Z, kt, lane);
#pragma unroll
  for (int j2 = 0; j2 < 16; ++j2) {
    f32x8 c = {};
#pragma unroll
    for (int kt = 0; kt < 4; ++kt)
      c = wmma_bf16(a4[kt], load_bfrag(W.w1 + (j2 * 4 + kt) * FRAG_ELEMS, lane), c);
    f32x8 h; float bb = W.b1v[j2];
#pragma unroll
    for (int r = 0; r < 8; ++r) h[r] = tanh_f(c[r] + bb);
    store_tile_bf16(W.act, ACT_STRIDE_H, j2, lane, h);  // a4 already in regs
  }
  // layer 2: K=256, N=256
  bf16x16 a8[8];
#pragma unroll
  for (int kt = 0; kt < 8; ++kt) a8[kt] = load_afrag(W.act, ACT_STRIDE_H, kt, lane);
#pragma unroll
  for (int j2 = 0; j2 < 16; ++j2) {
    f32x8 c = {};
#pragma unroll
    for (int kt = 0; kt < 8; ++kt)
      c = wmma_bf16(a8[kt], load_bfrag(W.w2 + (j2 * 8 + kt) * FRAG_ELEMS, lane), c);
    f32x8 h; float bb = W.b2v[j2];
#pragma unroll
    for (int r = 0; r < 8; ++r) h[r] = tanh_f(c[r] + bb);
    store_tile_bf16(W.act, ACT_STRIDE_H, j2, lane, h);  // a8 already in regs
  }
  // layer 3: K=256, N=128 (linear)
#pragma unroll
  for (int kt = 0; kt < 8; ++kt) a8[kt] = load_afrag(W.act, ACT_STRIDE_H, kt, lane);
#pragma unroll
  for (int j = 0; j < 8; ++j) {
    f32x8 c = {};
#pragma unroll
    for (int kt = 0; kt < 8; ++kt)
      c = wmma_bf16(a8[kt], load_bfrag(W.w3 + (j * 8 + kt) * FRAG_ELEMS, lane), c);
    float bb = W.b3v[j];
#pragma unroll
    for (int r = 0; r < 8; ++r) out[j][r] = c[r] + bb;
  }
}

__device__ __forceinline__ void write_z_rows(float* zdst, int rowbase, int lane,
                                             const f32x8 z[8]) {
  int mb = rowbase + ((lane >> 4) << 3);
  int nc = lane & 15;
#pragma unroll
  for (int j = 0; j < 8; ++j)
#pragma unroll
    for (int r = 0; r < 8; ++r)
      zdst[(size_t)(mb + r) * LATENT + j * 16 + nc] = z[j][r];
}

// Convert a PyTorch-style Linear weight [N,K] f32 into bf16 B-fragments.
__global__ __launch_bounds__(32) void prep_frags(const float* __restrict__ Wm,
                                                 __bf16* __restrict__ outf,
                                                 int K, int NKT) {
  const int frag = blockIdx.x;
  const int lane = threadIdx.x;
  const int j2 = frag / NKT;
  const int kt = frag - j2 * NKT;
  const int n  = j2 * 16 + (lane & 15);
  const int kb = kt * 32 + ((lane >> 4) << 3);
  __bf16* o = outf + (size_t)frag * FRAG_ELEMS;
  const float* src = Wm + (size_t)n * K + kb;
#pragma unroll
  for (int i = 0; i < 8; ++i) {
    o[lane * 8 + i]       = (__bf16)src[i];        // chunk0: k = kb..kb+7
    o[256 + lane * 8 + i] = (__bf16)src[16 + i];   // chunk1: k = kb+16..kb+23
  }
}

__global__ __launch_bounds__(128, 1)
void node_main(const float* __restrict__ traj, const __bf16* __restrict__ wsf,
               const float* __restrict__ be1, const float* __restrict__ be2,
               const float* __restrict__ wte, const float* __restrict__ bte,
               const float* __restrict__ b1,  const float* __restrict__ b2,
               const float* __restrict__ b3,  const float* __restrict__ bc1,
               const float* __restrict__ Wc2, const float* __restrict__ bc2,
               float* __restrict__ out) {
  extern __shared__ __bf16 smem[];
  const int tid  = threadIdx.x;
  const int lane = tid & 31;
  const int wave = tid >> 5;
  const int rowbase = blockIdx.x * ROWS_PER_BLK + wave * 16;
  const int ln16 = lane & 15;

  // stage W1/W2/W3 fragments (256 KB) into LDS once
  {
    const uint4* src = (const uint4*)wsf;
    uint4* dst = (uint4*)smem;
    for (int i = tid; i < LDS_W_BYTES / 16; i += (int)blockDim.x) dst[i] = src[i];
  }

  WaveCtx W;
  W.w1 = smem + OFF_W1; W.w2 = smem + OFF_W2; W.w3 = smem + OFF_W3;
  W.act = smem + (LDS_W_BYTES / 2) + wave * ACT_ELEMS;
  W.lane = lane;
#pragma unroll
  for (int j = 0; j < 16; ++j) { W.b1v[j] = b1[j * 16 + ln16]; W.b2v[j] = b2[j * 16 + ln16]; }
#pragma unroll
  for (int j = 0; j < 8; ++j) {
    W.b3v[j] = b3[j * 16 + ln16];
    W.wtev[j] = wte[j * 16 + ln16];
    W.btev[j] = bte[j * 16 + ln16];
  }

  const __bf16* we1f = wsf + OFF_WE1;
  const __bf16* we2f = wsf + OFF_WE2;
  const __bf16* wc1f = wsf + OFF_WC1;

  // ---------- encoder: z0 = (relu(x0 We1^T + be1)) We2^T + be2 ----------
  f32x8 z[8];
  {
    f32x8 c[16] = {};
#pragma unroll 1
    for (int kt = 0; kt < 24; ++kt) {      // K = 768
      bf16x16 a = load_afrag_gf32(traj, rowbase, ROW_F32, kt, lane);
#pragma unroll
      for (int j2 = 0; j2 < 16; ++j2)
        c[j2] = wmma_bf16(a, load_bfrag(we1f + (j2 * 24 + kt) * FRAG_ELEMS, lane), c[j2]);
    }
#pragma unroll
    for (int j2 = 0; j2 < 16; ++j2) {
      float bb = be1[j2 * 16 + ln16];
      f32x8 h;
#pragma unroll
      for (int r = 0; r < 8; ++r) h[r] = fmaxf(c[j2][r] + bb, 0.f);
      store_tile_bf16(W.act, ACT_STRIDE_H, j2, lane, h);
    }
    bf16x16 a8[8];
#pragma unroll
    for (int kt = 0; kt < 8; ++kt) a8[kt] = load_afrag(W.act, ACT_STRIDE_H, kt, lane);
#pragma unroll
    for (int j = 0; j < 8; ++j) {
      f32x8 cc = {};
#pragma unroll
      for (int kt = 0; kt < 8; ++kt)
        cc = wmma_bf16(a8[kt], load_bfrag(we2f + (j * 8 + kt) * FRAG_ELEMS, lane), cc);
      float bb = be2[j * 16 + ln16];
#pragma unroll
      for (int r = 0; r < 8; ++r) z[j][r] = cc[r] + bb;
    }
  }

  float* prob  = out;
  float* ztraj = out + BATCH;
  write_z_rows(ztraj, rowbase, lane, z);   // output row s=0

  __syncthreads();                          // weights resident in LDS

  // ---------- 54 RK4 (3/8 rule) steps ----------
#pragma unroll 1
  for (int n = 0; n < NSTEP; ++n) {
    const float t = (float)n * DT;
    f32x8 k[8], y[8], v[8], acc[8];
    ode_f(W, t, z, k);                                    // k1
#pragma unroll
    for (int j = 0; j < 8; ++j)
#pragma unroll
      for (int r = 0; r < 8; ++r) {
        y[j][r]   = z[j][r] + (DT / 3.f) * k[j][r];       // y2
        acc[j][r] = z[j][r] + (DT / 8.f) * k[j][r];
      }
    ode_f(W, t + DT / 3.f, y, k);                         // k2
#pragma unroll
    for (int j = 0; j < 8; ++j)
#pragma unroll
      for (int r = 0; r < 8; ++r) {
        float y2 = y[j][r], zz = z[j][r], k2 = k[j][r];
        v[j][r] = 3.f * y2 - 2.f * zz - DT * k2;          // z + dt(k1 - k2)
        y[j][r] = 2.f * zz - y2 + DT * k2;                // y3
        acc[j][r] += (3.f * DT / 8.f) * k2;
      }
    ode_f(W, t + 2.f * DT / 3.f, y, k);                   // k3
#pragma unroll
    for (int j = 0; j < 8; ++j)
#pragma unroll
      for (int r = 0; r < 8; ++r) {
        y[j][r] = v[j][r] + DT * k[j][r];                 // y4
        acc[j][r] += (3.f * DT / 8.f) * k[j][r];
      }
    ode_f(W, t + DT, y, k);                               // k4
#pragma unroll
    for (int j = 0; j < 8; ++j)
#pragma unroll
      for (int r = 0; r < 8; ++r) z[j][r] = acc[j][r] + (DT / 8.f) * k[j][r];

    if (n & 1)
      write_z_rows(ztraj + (size_t)((n >> 1) + 1) * BATCH * LATENT, rowbase, lane, z);
  }

  // ---------- classifier: sigmoid(relu(zL Wc1^T + bc1) Wc2^T + bc2) ----------
  {
#pragma unroll
    for (int j = 0; j < 8; ++j) store_tile_bf16(W.act, ACT_STRIDE_Z, j, lane, z[j]);
    bf16x16 a4[4];
#pragma unroll
    for (int kt = 0; kt < 4; ++kt) a4[kt] = load_afrag(W.act, ACT_STRIDE_Z, kt, lane);
    float part[8] = {0, 0, 0, 0, 0, 0, 0, 0};
#pragma unroll
    for (int j2 = 0; j2 < 16; ++j2) {
      f32x8 c = {};
#pragma unroll
      for (int kt = 0; kt < 4; ++kt)
        c = wmma_bf16(a4[kt], load_bfrag(wc1f + (j2 * 4 + kt) * FRAG_ELEMS, lane), c);
      float bb = bc1[j2 * 16 + ln16];
      float wv = Wc2[j2 * 16 + ln16];
#pragma unroll
      for (int r = 0; r < 8; ++r) part[r] += fmaxf(c[r] + bb, 0.f) * wv;
    }
    float* scr = (float*)W.act;
#pragma unroll
    for (int r = 0; r < 8; ++r) scr[lane * 8 + r] = part[r];
    if (lane < 16) {  // divergence only after all WMMAs
      int base = (lane < 8) ? 0 : 16;
      int rr = lane & 7;
      float s = 0.f;
#pragma unroll
      for (int l = 0; l < 16; ++l) s += scr[(base + l) * 8 + rr];
      float x = s + bc2[0];
      prob[rowbase + lane] = 0.5f * (1.f + tanh_f(0.5f * x));
    }
  }
}

extern "C" void kernel_launch(void* const* d_in, const int* in_sizes, int n_in,
                              void* d_out, int out_size, void* d_ws, size_t ws_size,
                              hipStream_t stream) {
  const float* traj = (const float*)d_in[0];
  const float* We1  = (const float*)d_in[1];
  const float* be1  = (const float*)d_in[2];
  const float* We2  = (const float*)d_in[3];
  const float* be2  = (const float*)d_in[4];
  const float* wte  = (const float*)d_in[5];
  const float* bte  = (const float*)d_in[6];
  const float* W1   = (const float*)d_in[7];
  const float* b1   = (const float*)d_in[8];
  const float* W2   = (const float*)d_in[9];
  const float* b2   = (const float*)d_in[10];
  const float* W3   = (const float*)d_in[11];
  const float* b3   = (const float*)d_in[12];
  const float* Wc1  = (const float*)d_in[13];
  const float* bc1  = (const float*)d_in[14];
  const float* Wc2  = (const float*)d_in[15];
  const float* bc2  = (const float*)d_in[16];
  __bf16* wsf = (__bf16*)d_ws;

  // one-time (per call) weight -> bf16 fragment conversion
  prep_frags<<<64,  32, 0, stream>>>(W1,  wsf + OFF_W1,  128, 4);
  prep_frags<<<128, 32, 0, stream>>>(W2,  wsf + OFF_W2,  256, 8);
  prep_frags<<<64,  32, 0, stream>>>(W3,  wsf + OFF_W3,  256, 8);
  prep_frags<<<384, 32, 0, stream>>>(We1, wsf + OFF_WE1, 768, 24);
  prep_frags<<<64,  32, 0, stream>>>(We2, wsf + OFF_WE2, 256, 8);
  prep_frags<<<64,  32, 0, stream>>>(Wc1, wsf + OFF_WC1, 128, 4);

  (void)hipFuncSetAttribute((const void*)node_main,
                            hipFuncAttributeMaxDynamicSharedMemorySize, LDS_TOTAL);
  node_main<<<BATCH / ROWS_PER_BLK, 128, LDS_TOTAL, stream>>>(
      traj, wsf, be1, be2, wte, bte, b1, b2, b3, bc1, Wc2, bc2, (float*)d_out);
}
